// Node_GCN_70523363000490
// MI455X (gfx1250) — matrix operand order repaired and checked by reading, not compiled, not run on hardware
//
#include <hip/hip_runtime.h>
#include <hip/hip_bf16.h>
#include <stdint.h>

#define N_NODES 100000
#define N_EDGES 1000000
#define KEEP 0.7f
#define INV_KEEP (1.0f/0.7f)

typedef __attribute__((ext_vector_type(16))) __bf16 v16bf;
typedef __attribute__((ext_vector_type(8)))  __bf16 v8bf;
typedef __attribute__((ext_vector_type(8)))  float  v8f;
typedef __attribute__((ext_vector_type(4)))  unsigned int v4u;
typedef __attribute__((ext_vector_type(8)))  int v8i_t;
typedef __attribute__((ext_vector_type(4)))  int v4i_t;

#if defined(__has_builtin)
#  if __has_builtin(__builtin_amdgcn_tensor_load_to_lds) && __has_builtin(__builtin_amdgcn_s_wait_tensorcnt)
#    define HAVE_TDM 1
#  endif
#endif
#ifndef HAVE_TDM
#  define HAVE_TDM 0
#endif

__device__ __forceinline__ unsigned short f2bf(float f) {
    union { float f; uint32_t u; } v; v.f = f;
    uint32_t u = v.u;
    uint32_t r = u + 0x7FFFu + ((u >> 16) & 1u);
    return (unsigned short)(r >> 16);
}

__device__ __forceinline__ v16bf cat8(v8bf lo, v8bf hi) {
    return __builtin_shufflevector(lo, hi, 0,1,2,3,4,5,6,7,8,9,10,11,12,13,14,15);
}

__device__ __forceinline__ uint32_t hash_u32(uint32_t a, uint32_t b, uint32_t seed) {
    uint32_t h = a * 0x9E3779B1u ^ b * 0x85EBCA77u ^ seed;
    h ^= h >> 16; h *= 0x7FEB352Du;
    h ^= h >> 15; h *= 0x846CA68Bu;
    h ^= h >> 16;
    return h;
}
__device__ __forceinline__ float hash_unit(uint32_t a, uint32_t b, uint32_t seed) {
    return (hash_u32(a, b, seed) >> 8) * (1.0f / 16777216.0f);
}

#if HAVE_TDM
// 1-D TDM copy: elems_u16 bf16 elements from global -> LDS (D# per CDNA5 ISA §8).
__device__ __forceinline__ void tdm_load_1d_to_lds(uint32_t lds_off, const void* gsrc,
                                                   uint32_t elems_u16) {
    uint64_t ga = (uint64_t)(uintptr_t)gsrc;
    v4u g0;
    g0[0] = 1u;                                               // count=1 (valid), user mode
    g0[1] = lds_off;                                          // lds_addr (bytes)
    g0[2] = (uint32_t)ga;                                     // global_addr[31:0]
    g0[3] = (uint32_t)((ga >> 32) & 0x01FFFFFFu) | (2u << 30); // global_addr[56:32] | type=2
    v8i_t g1;
    g1[0] = (int)(1u << 16);                                  // data_size=1 (2 bytes/elem)
    g1[1] = (int)((elems_u16 & 0xFFFFu) << 16);               // tensor_dim0[15:0]
    g1[2] = (int)(((elems_u16 >> 16) & 0xFFFFu) | (1u << 16)); // tensor_dim0[31:16], tensor_dim1=1
    g1[3] = (int)((elems_u16 & 0xFFFFu) << 16);               // tile_dim0 = elems
    g1[4] = 1;                                                // tile_dim1 = 1
    g1[5] = (int)elems_u16;                                   // tensor_dim0_stride[31:0]
    g1[6] = 0;                                                // stride hi / dim1_stride lo
    g1[7] = 0;
    v4i_t z4 = {0, 0, 0, 0};
#if defined(__clang_major__) && __clang_major__ >= 23
    v8i_t z8 = {0, 0, 0, 0, 0, 0, 0, 0};
    __builtin_amdgcn_tensor_load_to_lds(g0, g1, z4, z4, z8, 0);
#else
    __builtin_amdgcn_tensor_load_to_lds(g0, g1, z4, z4, 0);
#endif
}
#endif

// ---------------- utility kernels ----------------

__global__ void zero_kernel(float* p, int n) {
    int i = blockIdx.x * blockDim.x + threadIdx.x;
    if (i < n) p[i] = 0.0f;
}

// w[K][OUT] f32 (row-major) -> wt[OUT][K] bf16 (transposed)
__global__ void prep_w_kernel(const float* __restrict__ w, unsigned short* __restrict__ wt,
                              int K, int OUT) {
    int tid = blockIdx.x * blockDim.x + threadIdx.x;
    if (tid >= K * OUT) return;
    int k = tid / OUT, n = tid % OUT;
    wt[n * K + k] = f2bf(w[tid]);
}

// ---------------- PAE batchnorm statistics ----------------

__global__ void pae_stats_kernel(const float* __restrict__ en, const float* __restrict__ w1,
                                 const float* __restrict__ b1,
                                 float* __restrict__ sum, float* __restrict__ sq, int E) {
    __shared__ float buf[64 * 4];
    int tid = threadIdx.x;               // 256 threads: s = tid>>7, j = tid&127
    int s = tid >> 7, j = tid & 127;
    float wa = w1[j], wb = w1[128 + j], bb = b1[j];
    float acc = 0.0f, acc2 = 0.0f;
    int per = (E + gridDim.x - 1) / gridDim.x;
    int e0 = blockIdx.x * per;
    int e1 = e0 + per; if (e1 > E) e1 = E;
    for (int eb = e0; eb < e1; eb += 64) {
        __syncthreads();
        { int e = eb + (tid >> 2); if (e < e1) buf[tid] = en[e * 4 + (tid & 3)]; }
        if (eb + 64 < e1) __builtin_prefetch(&en[(eb + 64) * 4 + (tid & 3)], 0, 1);
        __syncthreads();
        int cnt = e1 - eb; if (cnt > 64) cnt = 64;
        for (int i = 0; i < cnt; ++i) {
            float x0 = buf[i * 4 + s * 2], x1 = buf[i * 4 + s * 2 + 1];
            float h = fmaxf(fmaf(x0, wa, fmaf(x1, wb, bb)), 0.0f);
            acc += h; acc2 += h * h;
        }
    }
    atomicAdd(&sum[tid], acc);
    atomicAdd(&sq[tid], acc2);
}

__global__ void pae_finalize_kernel(const float* __restrict__ sum, const float* __restrict__ sq,
                                    float* __restrict__ mean, float* __restrict__ invstd, int E) {
    int i = threadIdx.x;  // 256
    float m = sum[i] / (float)E;
    float v = sq[i] / (float)E - m * m;
    mean[i] = m;
    invstd[i] = rsqrtf(fmaxf(v, 0.0f) + 1e-5f);
}

// ---------------- PAE main: Linear2 via WMMA + cosine ----------------
// One wave (32 threads) per 16-edge tile. Hidden states (post BN+dropout) for
// both halves are built in LDS as bf16, then D = H @ W2t via v_wmma bf16,
// fusing bias, dot(h1,h2) and norms directly on the D fragments.

__global__ void pae_edge_kernel(const float* __restrict__ en, const float* __restrict__ w1,
                                const float* __restrict__ b1, const float* __restrict__ gamma,
                                const float* __restrict__ beta,
                                const unsigned short* __restrict__ w2t,
                                const float* __restrict__ b2, const float* __restrict__ mean,
                                const float* __restrict__ invstd, float* __restrict__ ew) {
    __shared__ __attribute__((aligned(32))) unsigned short Hs[2][16][128];
    const int lane = threadIdx.x;
    const int e0 = blockIdx.x * 16;

    // Phase 1: h = dropout(BN(relu(x @ w1 + b1))) for both halves, store bf16.
    for (int i = lane; i < 4096; i += 32) {
        int s = i >> 11;
        int m = (i >> 7) & 15;
        int j = i & 127;
        int e = e0 + m;
        float x0 = en[e * 4 + s * 2 + 0];
        float x1 = en[e * 4 + s * 2 + 1];
        float h = fmaxf(fmaf(x0, w1[j], fmaf(x1, w1[128 + j], b1[j])), 0.0f);
        int sj = s * 128 + j;
        h = fmaf((h - mean[sj]) * invstd[sj], gamma[j], beta[j]);
        h = (hash_unit((uint32_t)e, (uint32_t)sj, 0xC0FFEEu) < KEEP) ? h * INV_KEEP : 0.0f;
        Hs[s][m][j] = f2bf(h);
    }
    __syncthreads();

    const int m = lane & 15, hi = lane >> 4;
    float pd[8], p1[8], p2[8];
#pragma unroll
    for (int r = 0; r < 8; ++r) { pd[r] = 0.f; p1[r] = 0.f; p2[r] = 0.f; }

    for (int t = 0; t < 8; ++t) {              // 8 output tiles of 16 cols
        v8f acc0 = {0.f,0.f,0.f,0.f,0.f,0.f,0.f,0.f};
        v8f acc1 = {0.f,0.f,0.f,0.f,0.f,0.f,0.f,0.f};
        float bval = b2[t * 16 + m];
#pragma unroll
        for (int kk = 0; kk < 4; ++kk) {       // K = 128 in steps of 32
            int ka = kk * 32 + hi * 16;
            v8bf a0l = *(const v8bf*)&Hs[0][m][ka];
            v8bf a0h = *(const v8bf*)&Hs[0][m][ka + 8];
            v8bf a1l = *(const v8bf*)&Hs[1][m][ka];
            v8bf a1h = *(const v8bf*)&Hs[1][m][ka + 8];
            v16bf af0 = cat8(a0l, a0h);
            v16bf af1 = cat8(a1l, a1h);
            v16bf bf = *(const v16bf*)&w2t[(t * 16 + m) * 128 + kk * 32 + hi * 16];
            acc0 = __builtin_amdgcn_wmma_f32_16x16x32_bf16(false, af0, false, bf,
                                                           (short)0, acc0, false, false);
            acc1 = __builtin_amdgcn_wmma_f32_16x16x32_bf16(false, af1, false, bf,
                                                           (short)0, acc1, false, false);
        }
#pragma unroll
        for (int r = 0; r < 8; ++r) {
            float v1 = acc0[r] + bval, v2 = acc1[r] + bval;
            pd[r] = fmaf(v1, v2, pd[r]);
            p1[r] = fmaf(v1, v1, p1[r]);
            p2[r] = fmaf(v2, v2, p2[r]);
        }
    }
    // Reduce across the 16 lanes of each half (columns of the edge's row).
#pragma unroll
    for (int r = 0; r < 8; ++r) {
        for (int off = 8; off >= 1; off >>= 1) {
            pd[r] += __shfl_xor(pd[r], off, 32);
            p1[r] += __shfl_xor(p1[r], off, 32);
            p2[r] += __shfl_xor(p2[r], off, 32);
        }
    }
    if (m == 0) {
#pragma unroll
        for (int r = 0; r < 8; ++r) {
            float n1 = fmaxf(sqrtf(p1[r]), 1e-8f);
            float n2 = fmaxf(sqrtf(p2[r]), 1e-8f);
            float c = pd[r] / (n1 * n2);
            ew[e0 + hi * 8 + r] = (c + 1.0f) * 0.5f;
        }
    }
}

// ---------------- GCN degree / normalization ----------------

__global__ void deg_kernel(const int* __restrict__ col, const float* __restrict__ ew,
                           float* __restrict__ deg, int E) {
    int e = blockIdx.x * blockDim.x + threadIdx.x;
    if (e < E) atomicAdd(&deg[col[e]], ew[e]);
}

__global__ void dinv_kernel(const float* __restrict__ deg, float* __restrict__ dinv, int N) {
    int n = blockIdx.x * blockDim.x + threadIdx.x;
    if (n >= N) return;
    float d = deg[n] + 1.0f;  // + self-loop weight
    dinv[n] = (d > 0.0f) ? rsqrtf(d) : 0.0f;
}

// ---------------- generic WMMA GEMM: C = act(A[N x K] @ Wt[OUT x K]^T + b) ----
// One wave per 16-row tile. A converted f32->bf16 into LDS per tile. The bf16
// weight matrix is DMA'd into LDS by the Tensor Data Mover (overlapping the
// A-tile conversion), synchronized with s_wait_tensorcnt.

__global__ void gemm_wmma_kernel(const float* __restrict__ A,
                                 const unsigned short* __restrict__ Wt,
                                 const float* __restrict__ bias, float* __restrict__ C,
                                 int K, int OUT, int relu) {
    __shared__ __attribute__((aligned(32))) unsigned short tile[16 * 64];
#if HAVE_TDM
    __shared__ __attribute__((aligned(32))) unsigned short wlds[64 * 64];
    tdm_load_1d_to_lds((uint32_t)(uintptr_t)&wlds[0], Wt, (uint32_t)(OUT * K));
#endif
    const int lane = threadIdx.x;
    const int row0 = blockIdx.x * 16;
    const int elems = 16 * K;
    for (int i = lane; i < elems; i += 32) {
        int r = i / K, c = i % K;
        tile[r * K + c] = f2bf(A[(row0 + r) * K + c]);
    }
#if HAVE_TDM
    __builtin_amdgcn_s_wait_tensorcnt(0);
#endif
    __syncthreads();
    const int m = lane & 15, hi = lane >> 4;
    const int ktiles = K >> 5;
    for (int t = 0; t < (OUT >> 4); ++t) {
        v8f acc = {0.f,0.f,0.f,0.f,0.f,0.f,0.f,0.f};
        for (int kk = 0; kk < ktiles; ++kk) {
            int ka = kk * 32 + hi * 16;
            v8bf al = *(const v8bf*)&tile[m * K + ka];
            v8bf ah = *(const v8bf*)&tile[m * K + ka + 8];
            v16bf af = cat8(al, ah);
#if HAVE_TDM
            v16bf bf = *(const v16bf*)&wlds[(t * 16 + m) * K + kk * 32 + hi * 16];
#else
            v16bf bf = *(const v16bf*)&Wt[(t * 16 + m) * K + kk * 32 + hi * 16];
#endif
            acc = __builtin_amdgcn_wmma_f32_16x16x32_bf16(false, af, false, bf,
                                                          (short)0, acc, false, false);
        }
        float bval = bias ? bias[t * 16 + m] : 0.0f;
#pragma unroll
        for (int r = 0; r < 8; ++r) {
            int rr = r + hi * 8;
            float v = acc[r] + bval;
            if (relu) v = fmaxf(v, 0.0f);
            C[(row0 + rr) * OUT + t * 16 + m] = v;
        }
    }
}

// ---------------- edge aggregation (gather-scale-scatter) ----------------
// thread = (edge, feature); 4 edges per 256-thread block.

__global__ void aggregate_kernel(const int* __restrict__ row, const int* __restrict__ col,
                                 const float* __restrict__ ew, const float* __restrict__ dinv,
                                 const float* __restrict__ xw, float* __restrict__ acc, int E) {
    int tid = blockIdx.x * blockDim.x + threadIdx.x;
    int e = tid >> 6;
    if (e >= E) return;
    int j = tid & 63;
    int r = row[e], c = col[e];
    float norm = dinv[r] * ew[e] * dinv[c];
    atomicAdd(&acc[c * 64 + j], norm * xw[r * 64 + j]);
}

// self-loop contribution + bias + relu, in-place on the accumulator
__global__ void finish_conv_kernel(const float* __restrict__ xw, const float* __restrict__ dinv,
                                   const float* __restrict__ b, float* __restrict__ acc, int N) {
    int tid = blockIdx.x * blockDim.x + threadIdx.x;
    if (tid >= N * 64) return;
    int n = tid >> 6, j = tid & 63;
    float d = dinv[n];
    float v = acc[tid] + d * d * xw[tid] + b[j];
    acc[tid] = fmaxf(v, 0.0f);
}

// ---------------- head: dropout + Linear(32 -> 10) ----------------

__global__ void head_kernel(const float* __restrict__ hmid, const float* __restrict__ w,
                            const float* __restrict__ b, float* __restrict__ out, int N) {
    int n = blockIdx.x * blockDim.x + threadIdx.x;
    if (n >= N) return;
    float h[32];
#pragma unroll
    for (int j = 0; j < 32; ++j) {
        float v = hmid[n * 32 + j];
        h[j] = (hash_unit((uint32_t)n, (uint32_t)j, 0xD00D5EEDu) < KEEP) ? v * INV_KEEP : 0.0f;
    }
#pragma unroll
    for (int c = 0; c < 10; ++c) {
        float acc = b[c];
#pragma unroll
        for (int j = 0; j < 32; ++j) acc = fmaf(h[j], w[j * 10 + c], acc);
        out[n * 10 + c] = acc;
    }
}

// ---------------- host orchestration ----------------

extern "C" void kernel_launch(void* const* d_in, const int* in_sizes, int n_in,
                              void* d_out, int out_size, void* d_ws, size_t ws_size,
                              hipStream_t stream) {
    const float* x        = (const float*)d_in[0];
    const int*   ei       = (const int*)  d_in[1];
    const float* en       = (const float*)d_in[2];
    const float* pae_w1   = (const float*)d_in[3];
    const float* pae_b1   = (const float*)d_in[4];
    const float* pae_g    = (const float*)d_in[5];
    const float* pae_be   = (const float*)d_in[6];
    const float* pae_w2   = (const float*)d_in[7];
    const float* pae_b2   = (const float*)d_in[8];
    const float* conv1_w  = (const float*)d_in[9];
    const float* conv1_b  = (const float*)d_in[10];
    const float* conv2_w  = (const float*)d_in[11];
    const float* conv2_b  = (const float*)d_in[12];
    const float* lin1_w   = (const float*)d_in[13];
    const float* lin1_b   = (const float*)d_in[14];
    const float* lin2_w   = (const float*)d_in[15];
    const float* lin2_b   = (const float*)d_in[16];
    (void)in_sizes; (void)n_in; (void)out_size; (void)ws_size;

    const int* row = ei;
    const int* col = ei + N_EDGES;
    const int N = N_NODES, E = N_EDGES;

    char* wp = (char*)d_ws;
    auto alloc = [&](size_t bytes) -> char* {
        char* p = wp;
        wp += (bytes + 255) & ~(size_t)255;
        return p;
    };
    unsigned short* w2t  = (unsigned short*)alloc(128 * 128 * 2);
    unsigned short* c1wt = (unsigned short*)alloc(64 * 64 * 2);
    unsigned short* c2wt = (unsigned short*)alloc(64 * 64 * 2);
    unsigned short* l1wt = (unsigned short*)alloc(32 * 64 * 2);
    float* ssum  = (float*)alloc(256 * 4);
    float* ssq   = (float*)alloc(256 * 4);
    float* smean = (float*)alloc(256 * 4);
    float* sinv  = (float*)alloc(256 * 4);
    float* deg   = (float*)alloc((size_t)N * 4);
    float* dinv  = (float*)alloc((size_t)N * 4);
    float* ew    = (float*)alloc((size_t)E * 4);
    float* bufA  = (float*)alloc((size_t)N * 64 * 4);
    float* bufB  = (float*)alloc((size_t)N * 64 * 4);

    // weight prep (f32 -> transposed bf16)
    prep_w_kernel<<<(128 * 128 + 255) / 256, 256, 0, stream>>>(pae_w2, w2t, 128, 128);
    prep_w_kernel<<<(64 * 64 + 255) / 256, 256, 0, stream>>>(conv1_w, c1wt, 64, 64);
    prep_w_kernel<<<(64 * 64 + 255) / 256, 256, 0, stream>>>(conv2_w, c2wt, 64, 64);
    prep_w_kernel<<<(64 * 32 + 255) / 256, 256, 0, stream>>>(lin1_w, l1wt, 64, 32);

    // PAE: batchnorm stats, then WMMA edge kernel -> edge weights
    zero_kernel<<<1, 256, 0, stream>>>(ssum, 256);
    zero_kernel<<<1, 256, 0, stream>>>(ssq, 256);
    pae_stats_kernel<<<512, 256, 0, stream>>>(en, pae_w1, pae_b1, ssum, ssq, E);
    pae_finalize_kernel<<<1, 256, 0, stream>>>(ssum, ssq, smean, sinv, E);
    pae_edge_kernel<<<E / 16, 32, 0, stream>>>(en, pae_w1, pae_b1, pae_g, pae_be,
                                               w2t, pae_b2, smean, sinv, ew);

    // symmetric normalization
    zero_kernel<<<(N + 255) / 256, 256, 0, stream>>>(deg, N);
    deg_kernel<<<(E + 255) / 256, 256, 0, stream>>>(col, ew, deg, E);
    dinv_kernel<<<(N + 255) / 256, 256, 0, stream>>>(deg, dinv, N);

    // conv1
    gemm_wmma_kernel<<<N / 16, 32, 0, stream>>>(x, c1wt, nullptr, bufA, 64, 64, 0);
    zero_kernel<<<(N * 64 + 255) / 256, 256, 0, stream>>>(bufB, N * 64);
    aggregate_kernel<<<(E * 64) / 256, 256, 0, stream>>>(row, col, ew, dinv, bufA, bufB, E);
    finish_conv_kernel<<<(N * 64 + 255) / 256, 256, 0, stream>>>(bufA, dinv, conv1_b, bufB, N);

    // conv2
    gemm_wmma_kernel<<<N / 16, 32, 0, stream>>>(bufB, c2wt, nullptr, bufA, 64, 64, 0);
    zero_kernel<<<(N * 64 + 255) / 256, 256, 0, stream>>>(bufB, N * 64);
    aggregate_kernel<<<(E * 64) / 256, 256, 0, stream>>>(row, col, ew, dinv, bufA, bufB, E);
    finish_conv_kernel<<<(N * 64 + 255) / 256, 256, 0, stream>>>(bufA, dinv, conv2_b, bufB, N);

    // head: lin1 (relu) via WMMA, then dropout + lin2
    gemm_wmma_kernel<<<N / 16, 32, 0, stream>>>(bufB, l1wt, lin1_b, bufA, 64, 32, 1);
    head_kernel<<<(N + 255) / 256, 256, 0, stream>>>(bufA, lin2_w, lin2_b, (float*)d_out, N);
}
